// Combined_2_54013508714662
// MI455X (gfx1250) — matrix operand order repaired and checked by reading, not compiled, not run on hardware
//
#include <hip/hip_runtime.h>
#include <hip/hip_bf16.h>

typedef __attribute__((ext_vector_type(16))) _Float16 v16h;
typedef __attribute__((ext_vector_type(8)))  _Float16 v8h;
typedef __attribute__((ext_vector_type(8)))  float    v8f;

#define C_IN   64
#define KNN    16
#define NCLS   40
#define NC_PAD 48

// ---------------------------------------------------------------------------
// Kernel 1: CPE depthwise conv1d (kernel 3, pad 1, groups=C) + residual
// ---------------------------------------------------------------------------
__global__ __launch_bounds__(256)
void cpe_kernel(const float* __restrict__ x,
                const float* __restrict__ cpe_w,   // [C,3]
                const float* __restrict__ cpe_b,   // [C]
                float* __restrict__ h, int N)
{
    __shared__ float sw[C_IN * 3];
    __shared__ float sb[C_IN];
    if (threadIdx.x < C_IN * 3) sw[threadIdx.x] = cpe_w[threadIdx.x];
    if (threadIdx.x < C_IN)     sb[threadIdx.x] = cpe_b[threadIdx.x];
    __syncthreads();

    int n = blockIdx.x * blockDim.x + threadIdx.x;
    if (n >= N) return;
    const float4* x4 = reinterpret_cast<const float4*>(x);
    float4*       h4 = reinterpret_cast<float4*>(h);
    const bool hm = (n > 0), hp = (n < N - 1);
    const float4 z4 = make_float4(0.f, 0.f, 0.f, 0.f);

#pragma unroll
    for (int q = 0; q < C_IN / 4; ++q) {
        float4 xc = x4[n * 16 + q];
        float4 xm = hm ? x4[(n - 1) * 16 + q] : z4;
        float4 xp = hp ? x4[(n + 1) * 16 + q] : z4;
        int c = q * 4;
        float4 o;
        o.x = xc.x + xm.x * sw[(c+0)*3+0] + xc.x * sw[(c+0)*3+1] + xp.x * sw[(c+0)*3+2] + sb[c+0];
        o.y = xc.y + xm.y * sw[(c+1)*3+0] + xc.y * sw[(c+1)*3+1] + xp.y * sw[(c+1)*3+2] + sb[c+1];
        o.z = xc.z + xm.z * sw[(c+2)*3+0] + xc.z * sw[(c+2)*3+1] + xp.z * sw[(c+2)*3+2] + sb[c+2];
        o.w = xc.w + xm.w * sw[(c+3)*3+0] + xc.w * sw[(c+3)*3+1] + xp.w * sw[(c+3)*3+2] + sb[c+3];
        h4[n * 16 + q] = o;
    }
}

// ---------------------------------------------------------------------------
// Kernel 2 (fused): gather/max-rel -> grapher WMMA -> residual -> head WMMA
//                   -> log_softmax.
// Block = 256 threads = 8 waves; each wave owns a 16-node tile.
// Weights live in LDS *transposed* (Wt[col][k], f16) so every WMMA B fragment
// (b[e<8] = W[K0+koff+e][col], b[e>=8] = W[K0+16+koff+e-8][col]) is two 16B LDS loads (2x ds_load_b128).
// h2 tile lives in LDS as f16 row-major so head A fragments are b128 loads.
// Neighbor indices are staged per block via global_load_async_to_lds_b128.
// ---------------------------------------------------------------------------
__global__ __launch_bounds__(256)
void graph_head_kernel(const float* __restrict__ h,
                       const int*   __restrict__ nbr,   // [N,K]
                       const float* __restrict__ g_w,   // [2C, C]
                       const float* __restrict__ g_b,   // [C]
                       const float* __restrict__ o_w,   // [C, 40]
                       const float* __restrict__ o_b,   // [40]
                       float* __restrict__ out, int N)
{
    __shared__ __align__(32) _Float16 sGwT[C_IN * 128];     // 16 KB  [col][k]
    __shared__ __align__(32) _Float16 sOwT[NC_PAD * C_IN];  //  6 KB  [col][k]
    __shared__ __align__(32) _Float16 sH2h[8][16 * C_IN];   // 16 KB  per-wave h2 (f16)
    __shared__ __align__(16) int      sNbr[128 * KNN];      //  8 KB
    __shared__ float sGb[C_IN];
    __shared__ float sOb[NC_PAD];

    const int tid = threadIdx.x;

    // ---- async-stage this block's neighbor indices: 8KB, 32B per thread ----
    {
        unsigned ldsOff = (unsigned)(uintptr_t)(&sNbr[0]) + (unsigned)tid * 32u;
        unsigned gOff   = (unsigned)tid * 32u;
        const int* gsrc = nbr + (size_t)blockIdx.x * (128 * KNN);
        asm volatile(
            "global_load_async_to_lds_b128 %0, %1, %2\n\t"
            "global_load_async_to_lds_b128 %0, %1, %2 offset:16"
            :: "v"(ldsOff), "v"(gOff), "s"(gsrc) : "memory");
    }

    // ---- stage weights transposed as f16 ----
    for (int i = tid; i < 128 * C_IN; i += 256) {           // i = k*64 + c
        int k = i >> 6, c = i & 63;
        sGwT[c * 128 + k] = (_Float16)g_w[i];
    }
    for (int i = tid; i < C_IN * NC_PAD; i += 256) {        // i = k*48 + c
        int k = i / NC_PAD, c = i % NC_PAD;
        sOwT[c * C_IN + k] = (_Float16)((c < NCLS) ? o_w[k * NCLS + c] : 0.0f);
    }
    if (tid < C_IN)   sGb[tid] = g_b[tid];
    if (tid < NC_PAD) sOb[tid] = (tid < NCLS) ? o_b[tid] : -1.0e30f;

    asm volatile("s_wait_asynccnt 0" ::: "memory");
    __syncthreads();

    const int wave  = tid >> 5;
    const int lane  = tid & 31;
    const int lhalf = lane >> 4;          // 0 / 1
    const int lrow  = lane & 15;          // A-row M  /  D-column N
    const int kb    = lhalf * 8;          // A-fragment K sub-offset
    const int koffB = lhalf * 8;          // B-fragment K sub-offset
    const int drow  = lhalf * 8;          // D-fragment row offset
    const int tileBase = blockIdx.x * 128 + wave * 16;
    const int node     = tileBase + lrow;
    const int nloc     = wave * 16 + lrow;       // node index within block

    // ---- gather phase: own h channels + max over K neighbors (L2-resident) ----
    float hreg[4][8];
    float mreg[4][8];
    const float4* h4 = reinterpret_cast<const float4*>(h);
#pragma unroll
    for (int cc = 0; cc < 4; ++cc) {
        int c0 = cc * 16 + kb;
        float4 a0 = h4[node * 16 + (c0 >> 2)];
        float4 a1 = h4[node * 16 + (c0 >> 2) + 1];
        hreg[cc][0]=a0.x; hreg[cc][1]=a0.y; hreg[cc][2]=a0.z; hreg[cc][3]=a0.w;
        hreg[cc][4]=a1.x; hreg[cc][5]=a1.y; hreg[cc][6]=a1.z; hreg[cc][7]=a1.w;
#pragma unroll
        for (int j = 0; j < 8; ++j) mreg[cc][j] = -3.0e38f;
    }
#pragma unroll 4
    for (int k = 0; k < KNN; ++k) {
        int idx = sNbr[nloc * KNN + k];
#pragma unroll
        for (int cc = 0; cc < 4; ++cc) {
            int c0 = cc * 16 + kb;
            float4 b0 = h4[idx * 16 + (c0 >> 2)];
            float4 b1 = h4[idx * 16 + (c0 >> 2) + 1];
            mreg[cc][0]=fmaxf(mreg[cc][0],b0.x); mreg[cc][1]=fmaxf(mreg[cc][1],b0.y);
            mreg[cc][2]=fmaxf(mreg[cc][2],b0.z); mreg[cc][3]=fmaxf(mreg[cc][3],b0.w);
            mreg[cc][4]=fmaxf(mreg[cc][4],b1.x); mreg[cc][5]=fmaxf(mreg[cc][5],b1.y);
            mreg[cc][6]=fmaxf(mreg[cc][6],b1.z); mreg[cc][7]=fmaxf(mreg[cc][7],b1.w);
        }
    }
#pragma unroll
    for (int cc = 0; cc < 4; ++cc)
#pragma unroll
        for (int j = 0; j < 8; ++j) mreg[cc][j] -= hreg[cc][j];   // rel = max - h

    // ---- grapher WMMA: h2[16,64] = feat[16,128] @ g_w[128,64] ----
    const v8f vzero = {0.f,0.f,0.f,0.f,0.f,0.f,0.f,0.f};
    v8f acc[4];
#pragma unroll
    for (int t = 0; t < 4; ++t) acc[t] = vzero;

#pragma unroll
    for (int ks = 0; ks < 4; ++ks) {      // K0 = 32*ks ; feat = [h | rel]
        v16h a;
#pragma unroll
        for (int j = 0; j < 8; ++j) {
            float lo = (ks == 0) ? hreg[0][j] : (ks == 1) ? hreg[2][j]
                     : (ks == 2) ? mreg[0][j] : mreg[2][j];
            float hi = (ks == 0) ? hreg[1][j] : (ks == 1) ? hreg[3][j]
                     : (ks == 2) ? mreg[1][j] : mreg[3][j];
            a[j]     = (_Float16)lo;      // K = K0 + kb + j
            a[8 + j] = (_Float16)hi;      // K = K0 + 16 + kb + j
        }
#pragma unroll
        for (int t = 0; t < 4; ++t) {
            // two 8-half runs: Wt[col][K0+koffB .. +7], Wt[col][K0+16+koffB .. +7]
            const v8h b0 = *reinterpret_cast<const v8h*>(
                &sGwT[(t * 16 + lrow) * 128 + ks * 32 + koffB]);
            const v8h b1 = *reinterpret_cast<const v8h*>(
                &sGwT[(t * 16 + lrow) * 128 + ks * 32 + 16 + koffB]);
            const v16h b = __builtin_shufflevector(b0, b1,
                 0,1,2,3,4,5,6,7,8,9,10,11,12,13,14,15);
            acc[t] = __builtin_amdgcn_wmma_f32_16x16x32_f16(
                false, a, false, b, (short)0, acc[t], false, false);
        }
    }

    // residual + bias -> f16 h2 tile in LDS (row-major, A-fragment friendly)
#pragma unroll
    for (int t = 0; t < 4; ++t)
#pragma unroll
        for (int v = 0; v < 8; ++v) {
            int row  = v + drow;
            int colg = t * 16 + lrow;
            float val = acc[t][v] + sGb[colg] + h[(tileBase + row) * C_IN + colg];
            sH2h[wave][row * C_IN + colg] = (_Float16)val;
        }
    __syncthreads();

    // ---- head WMMA: logits[16,48] = h2[16,64] @ o_w[64,48] ----
    v8f lg[3];
#pragma unroll
    for (int t = 0; t < 3; ++t) lg[t] = vzero;

#pragma unroll
    for (int ks = 0; ks < 2; ++ks) {      // K0 = 32*ks
        const v8h lo8 = *reinterpret_cast<const v8h*>(
            &sH2h[wave][lrow * C_IN + ks * 32 + kb]);
        const v8h hi8 = *reinterpret_cast<const v8h*>(
            &sH2h[wave][lrow * C_IN + ks * 32 + 16 + kb]);
        v16h a = __builtin_shufflevector(lo8, hi8,
                 0,1,2,3,4,5,6,7,8,9,10,11,12,13,14,15);
#pragma unroll
        for (int t = 0; t < 3; ++t) {
            const v8h b0 = *reinterpret_cast<const v8h*>(
                &sOwT[(t * 16 + lrow) * C_IN + ks * 32 + koffB]);
            const v8h b1 = *reinterpret_cast<const v8h*>(
                &sOwT[(t * 16 + lrow) * C_IN + ks * 32 + 16 + koffB]);
            const v16h b = __builtin_shufflevector(b0, b1,
                 0,1,2,3,4,5,6,7,8,9,10,11,12,13,14,15);
            lg[t] = __builtin_amdgcn_wmma_f32_16x16x32_f16(
                false, a, false, b, (short)0, lg[t], false, false);
        }
    }

    // ---- log_softmax over 40 classes; pad classes carry -1e30 bias -> exp=0 ----
#pragma unroll
    for (int v = 0; v < 8; ++v) {
        float l0 = lg[0][v] + sOb[lrow];
        float l1 = lg[1][v] + sOb[16 + lrow];
        float l2 = lg[2][v] + sOb[32 + lrow];
        float m = fmaxf(l0, fmaxf(l1, l2));
#pragma unroll
        for (int d = 1; d < 16; d <<= 1) m = fmaxf(m, __shfl_xor(m, d, 32));
        float s = expf(l0 - m) + expf(l1 - m) + expf(l2 - m);
#pragma unroll
        for (int d = 1; d < 16; d <<= 1) s += __shfl_xor(s, d, 32);
        float lse = m + logf(s);
        int nrow = tileBase + v + drow;
        out[nrow * NCLS + lrow]      = l0 - lse;
        out[nrow * NCLS + 16 + lrow] = l1 - lse;
        if (lrow < 8)
            out[nrow * NCLS + 32 + lrow] = l2 - lse;
    }
}

// ---------------------------------------------------------------------------
extern "C" void kernel_launch(void* const* d_in, const int* in_sizes, int n_in,
                              void* d_out, int out_size, void* d_ws, size_t ws_size,
                              hipStream_t stream)
{
    const float* x     = (const float*)d_in[0];
    const int*   nbr   = (const int*)  d_in[1];
    const float* cpe_w = (const float*)d_in[2];
    const float* cpe_b = (const float*)d_in[3];
    const float* g_w   = (const float*)d_in[4];
    const float* g_b   = (const float*)d_in[5];
    const float* o_w   = (const float*)d_in[6];
    const float* o_b   = (const float*)d_in[7];
    float* out = (float*)d_out;
    float* h   = (float*)d_ws;                // N*64 floats = 64 MB scratch

    const int N = in_sizes[0] / C_IN;         // 262144

    cpe_kernel<<<(N + 255) / 256, 256, 0, stream>>>(x, cpe_w, cpe_b, h, N);
    graph_head_kernel<<<N / 128, 256, 0, stream>>>(h, nbr, g_w, g_b, o_w, o_b, out, N);
}
